// MLP_32160715113120
// MI455X (gfx1250) — compile-verified, matrix operand order checked by reading, not run on hardware
//
#include <hip/hip_runtime.h>

// LIIF-style local-implicit MLP, fused WMMA GEMM kernel for gfx1250 (MI455X).
// B=4, C=64, H=W=64, Q=65536. MLP: 132 -> 256 x4 (ReLU) -> 3.
// 32 queries x 4 corners = 128 GEMM rows per workgroup (8 wave32).
// f16 activations/weights, f32 accumulation via v_wmma_f32_16x16x32_f16.
// Hidden layers: each wave owns a 4x4 block of 16x16 tiles (wave grid 2x4 over
// the 8 m-tiles x 16 n-tiles) so each weight fragment fetched from L2 feeds
// 4 WMMAs from registers (4x less L2 traffic than 1x16 mapping).

typedef __attribute__((ext_vector_type(16))) _Float16 v16h;
typedef __attribute__((ext_vector_type(8)))  _Float16 v8h;
typedef __attribute__((ext_vector_type(8)))  float    v8f;

#define Bsz 4
#define Cch 64
#define Hh  64
#define Ww  64
#define Qq  65536
#define STRIDE 264      // LDS row stride in halves (256 + 8 pad, 528B, bank-skewed)
#define ROWS 128        // rows per workgroup = 32 queries * 4 corners
#define QPB 32          // queries per block

// Fragment-table offsets (in fragments of 32 lanes x 16 halves = 512 halves)
//  L0: 16 ntiles * 5 ksteps  = 80   frags  @ 0
//  L1: 16 * 8               = 128  frags  @ 80
//  L2: 128                          @ 208
//  L3: 128                          @ 336
//  L4: 1 * 8                = 8     frags  @ 464   (N padded 3->16)
#define NFRAGS 472

union V16U { v16h v; struct { v8h lo, hi; } p; };

// ---------------- weight packing: fp32 [N][K] -> f16 WMMA B-fragments --------
// B column n of (X @ W^T) is row n of W, so B-fragment element (lane,h) maps to
//   n = ntile*16 + (lane&15),  k = kstep*32 + h + 8*(h>=8) + 8*(lane>=16)
__global__ __launch_bounds__(256) void pack_weights_kernel(
    const float* __restrict__ w0, const float* __restrict__ w1,
    const float* __restrict__ w2, const float* __restrict__ w3,
    const float* __restrict__ w4, _Float16* __restrict__ packed)
{
  int t = blockIdx.x * blockDim.x + threadIdx.x;
  if (t >= NFRAGS * 32) return;
  int frag = t >> 5;
  int lane = t & 31;
  const float* W; int Nr, Kr, KS, fl;
  if (frag < 80)       { W = w0; Nr = 256; Kr = 132; KS = 5; fl = frag;       }
  else if (frag < 208) { W = w1; Nr = 256; Kr = 256; KS = 8; fl = frag - 80;  }
  else if (frag < 336) { W = w2; Nr = 256; Kr = 256; KS = 8; fl = frag - 208; }
  else if (frag < 464) { W = w3; Nr = 256; Kr = 256; KS = 8; fl = frag - 336; }
  else                 { W = w4; Nr = 3;   Kr = 256; KS = 8; fl = frag - 464; }
  int nt = fl / KS, ks = fl - nt * KS;
  int n  = nt * 16 + (lane & 15);
  int kx = (lane & 16) ? 8 : 0;
  _Float16* dst = packed + ((size_t)frag * 32 + lane) * 16;
#pragma unroll
  for (int h = 0; h < 16; ++h) {
    int k = ks * 32 + h + ((h >= 8) ? 8 : 0) + kx;
    float v = (n < Nr && k < Kr) ? W[(size_t)n * Kr + k] : 0.0f;
    dst[h] = (_Float16)v;
  }
}

// ---------------- fragment loads ---------------------------------------------
__device__ __forceinline__ v16h load_a_frag(const _Float16* __restrict__ buf,
                                            int row, int kbase) {
  // A fragment (16x32, f16): lane l holds row l&15; K chunks [kb..kb+7],[kb+16..kb+23]
  const _Float16* p = buf + row * STRIDE + kbase;
  V16U u;
  u.p.lo = *(const v8h*)(p);
  u.p.hi = *(const v8h*)(p + 16);
  return u.v;
}

__device__ __forceinline__ v16h load_b_frag(const _Float16* __restrict__ wf,
                                            int fragIdx, int lane) {
  const _Float16* p = wf + ((size_t)fragIdx * 32 + lane) * 16; // 32B/lane, coalesced
  V16U u;
  u.p.lo = *(const v8h*)(p);
  u.p.hi = *(const v8h*)(p + 8);
  return u.v;
}

// ---------------- hidden MLP layer: wave = 4 m-tiles x 4 n-tiles -------------
template <int KSTEPS, bool RELU>
__device__ __forceinline__ void mlp_layer_hidden(
    const _Float16* __restrict__ inb, _Float16* __restrict__ outb,
    const _Float16* __restrict__ wf, const float* __restrict__ biasL,
    int wave, int lane)
{
  const int mg = wave >> 2;            // 0..1: m-tiles mg*4 .. mg*4+3
  const int ng = wave & 3;             // 0..3: n-tiles ng*4 .. ng*4+3
  const int col16 = lane & 15;
  const int rhalf = (lane & 16) ? 8 : 0;
  v8f acc[4][4];                       // [m][n], 128 VGPRs
#pragma unroll
  for (int n = 0; n < 4; ++n) {
    float bv = biasL[(ng * 4 + n) * 16 + col16];  // bias folded into C init
    v8f a;
#pragma unroll
    for (int i = 0; i < 8; ++i) a[i] = bv;
#pragma unroll
    for (int m = 0; m < 4; ++m) acc[m][n] = a;
  }
#pragma unroll
  for (int ks = 0; ks < KSTEPS; ++ks) {
    v16h afrag[4];
#pragma unroll
    for (int m = 0; m < 4; ++m)
      afrag[m] = load_a_frag(inb, (mg * 4 + m) * 16 + col16, ks * 32 + rhalf);
#pragma unroll
    for (int n = 0; n < 4; ++n) {
      v16h bfrag = load_b_frag(wf, (ng * 4 + n) * KSTEPS + ks, lane);
#pragma unroll
      for (int m = 0; m < 4; ++m)      // reuse B fragment across 4 m-tiles
        acc[m][n] = __builtin_amdgcn_wmma_f32_16x16x32_f16(
            false, afrag[m], false, bfrag, (short)0, acc[m][n], false, false);
    }
  }
  // C/D layout: VGPR v, lane l -> row = mtile*16 + v + 8*(l>=16), col = l&15
#pragma unroll
  for (int m = 0; m < 4; ++m) {
#pragma unroll
    for (int n = 0; n < 4; ++n) {
#pragma unroll
      for (int v = 0; v < 8; ++v) {
        float x = acc[m][n][v];
        if (RELU) x = fmaxf(x, 0.0f);
        int row = (mg * 4 + m) * 16 + v + rhalf;
        int col = (ng * 4 + n) * 16 + col16;
        outb[row * STRIDE + col] = (_Float16)x;
      }
    }
  }
}

// ---------------- final layer: 256 -> 16 (3 valid), wave = 1 m-tile ----------
__device__ __forceinline__ void mlp_layer_final(
    const _Float16* __restrict__ inb, const _Float16* __restrict__ wf,
    const float* __restrict__ biasL, float* __restrict__ predOut,
    int m_tile, int lane)
{
  const int col16 = lane & 15;
  const int rhalf = (lane & 16) ? 8 : 0;
  float bv = biasL[col16];
  v8f acc;
#pragma unroll
  for (int i = 0; i < 8; ++i) acc[i] = bv;
#pragma unroll
  for (int ks = 0; ks < 8; ++ks) {
    v16h afrag = load_a_frag(inb, m_tile * 16 + col16, ks * 32 + rhalf);
    v16h bfrag = load_b_frag(wf, ks, lane);
    acc = __builtin_amdgcn_wmma_f32_16x16x32_f16(
        false, afrag, false, bfrag, (short)0, acc, false, false);
  }
#pragma unroll
  for (int v = 0; v < 8; ++v) {
    int row = m_tile * 16 + v + rhalf;
    if (col16 < 4) predOut[row * 4 + col16] = acc[v];
  }
}

// ---------------- main fused kernel ------------------------------------------
__global__ __launch_bounds__(256) void liif_mlp_kernel(
    const float* __restrict__ inp1, const float* __restrict__ inp2,
    const float* __restrict__ coord, const float* __restrict__ cell1,
    const float* __restrict__ b0, const float* __restrict__ b1,
    const float* __restrict__ b2, const float* __restrict__ b3,
    const float* __restrict__ b4, const _Float16* __restrict__ packed,
    float* __restrict__ out)
{
  __shared__ __align__(16) _Float16 bufA[ROWS * STRIDE];
  __shared__ __align__(16) _Float16 bufB[ROWS * STRIDE];
  __shared__ float predLDS[ROWS * 4];
  __shared__ float areaLDS[ROWS];
  __shared__ float biasLDS[1040];

  const int tid = threadIdx.x;
  const int wg  = blockIdx.x;

  // preload biases (L4 padded to 16 with zeros)
  for (int i = tid; i < 1040; i += 256) {
    float v;
    if      (i < 256)  v = b0[i];
    else if (i < 512)  v = b1[i - 256];
    else if (i < 768)  v = b2[i - 512];
    else if (i < 1024) v = b3[i - 768];
    else { int o = i - 1024; v = (o < 3) ? b4[o] : 0.0f; }
    biasLDS[i] = v;
  }

  // ---- gather: build X[128 x 160] f16 + per-corner areas --------------------
  {
    int r = tid >> 1;            // row 0..127  (row = 4*localQ + corner)
    int halfsel = tid & 1;       // 2 threads per row, 32 channels each
    int gq = wg * QPB + (r >> 2);
    int j  = r & 3;              // corner: (vx,vy) = ((j&2)?+1:-1, (j&1)?+1:-1)
    int b  = gq >> 16;           // gq / Q (Q = 65536)
    float c0 = coord[(size_t)gq * 2 + 0];
    float c1 = coord[(size_t)gq * 2 + 1];
    float vx = (j & 2) ? 1.0f : -1.0f;
    float vy = (j & 1) ? 1.0f : -1.0f;
    const float lo = -1.0f + 1e-6f, hi = 1.0f - 1e-6f;
    float cc0 = fminf(fmaxf(c0 + vx * (1.0f / Hh) + 1e-6f, lo), hi);
    float cc1 = fminf(fmaxf(c1 + vy * (1.0f / Ww) + 1e-6f, lo), hi);
    int ih = (int)rintf((cc0 + 1.0f) * (0.5f * Hh) - 0.5f);  // round-to-even = jnp.round
    int iw = (int)rintf((cc1 + 1.0f) * (0.5f * Ww) - 0.5f);
    ih = ::min(::max(ih, 0), Hh - 1);
    iw = ::min(::max(iw, 0), Ww - 1);
    int flat = ih * Ww + iw;
    const float* f1 = inp1 + (size_t)b * Cch * Hh * Ww + flat;
    const float* f2 = inp2 + (size_t)b * Cch * Hh * Ww + flat;
    _Float16* xr = bufA + r * STRIDE;
#pragma unroll 4
    for (int c = 0; c < 32; ++c) {
      int ch = halfsel * 32 + c;
      xr[ch]      = (_Float16)f1[(size_t)ch * (Hh * Ww)];   // feat1 -> cols 0..63
      xr[68 + ch] = (_Float16)f2[(size_t)ch * (Hh * Ww)];   // feat2 -> cols 68..131
    }
    if (halfsel == 0) {
      float qc0 = -1.0f + (2.0f * ih + 1.0f) * (1.0f / Hh);
      float qc1 = -1.0f + (2.0f * iw + 1.0f) * (1.0f / Ww);
      float rel0 = (c0 - qc0) * Hh;
      float rel1 = (c1 - qc1) * Ww;
      xr[64] = (_Float16)rel0;
      xr[65] = (_Float16)rel1;
      xr[66] = (_Float16)(cell1[(size_t)gq * 2 + 0] * Hh);
      xr[67] = (_Float16)(cell1[(size_t)gq * 2 + 1] * Ww);
#pragma unroll
      for (int k = 132; k < 160; ++k) xr[k] = (_Float16)0.0f;  // pad K 132 -> 160
      areaLDS[r] = fabsf(rel0 * rel1) + 1e-9f;
    }
  }
  __syncthreads();

  // ---- MLP: 4 hidden GEMM layers (4x4 tile blocks per wave) + final ---------
  const int wave = tid >> 5;
  const int lane = tid & 31;
  mlp_layer_hidden<5, true>(bufA, bufB, packed + (size_t)0   * 512, biasLDS + 0,   wave, lane);
  __syncthreads();
  mlp_layer_hidden<8, true>(bufB, bufA, packed + (size_t)80  * 512, biasLDS + 256, wave, lane);
  __syncthreads();
  mlp_layer_hidden<8, true>(bufA, bufB, packed + (size_t)208 * 512, biasLDS + 512, wave, lane);
  __syncthreads();
  mlp_layer_hidden<8, true>(bufB, bufA, packed + (size_t)336 * 512, biasLDS + 768, wave, lane);
  __syncthreads();
  mlp_layer_final(bufA, packed + (size_t)464 * 512, biasLDS + 1024, predLDS, wave, lane);
  __syncthreads();

  // ---- local ensemble: areas swapped 0<->3, 1<->2 ---------------------------
  if (tid < QPB * 3) {
    int lq = tid / 3, o = tid - lq * 3;
    int rb = lq * 4;
    float a0 = areaLDS[rb + 0], a1 = areaLDS[rb + 1];
    float a2 = areaLDS[rb + 2], a3 = areaLDS[rb + 3];
    float tot = a0 + a1 + a2 + a3;
    float s = predLDS[(rb + 0) * 4 + o] * a3 + predLDS[(rb + 1) * 4 + o] * a2
            + predLDS[(rb + 2) * 4 + o] * a1 + predLDS[(rb + 3) * 4 + o] * a0;
    out[((size_t)wg * QPB + lq) * 3 + o] = s / tot;
  }
}

extern "C" void kernel_launch(void* const* d_in, const int* in_sizes, int n_in,
                              void* d_out, int out_size, void* d_ws, size_t ws_size,
                              hipStream_t stream) {
  const float* inp1  = (const float*)d_in[0];
  const float* inp2  = (const float*)d_in[1];
  const float* coord = (const float*)d_in[2];
  const float* cell1 = (const float*)d_in[3];
  const float* w0 = (const float*)d_in[4];
  const float* b0 = (const float*)d_in[5];
  const float* w1 = (const float*)d_in[6];
  const float* b1 = (const float*)d_in[7];
  const float* w2 = (const float*)d_in[8];
  const float* b2 = (const float*)d_in[9];
  const float* w3 = (const float*)d_in[10];
  const float* b3 = (const float*)d_in[11];
  const float* w4 = (const float*)d_in[12];
  const float* b4 = (const float*)d_in[13];
  _Float16* packed = (_Float16*)d_ws;   // ~472 KB; L2-resident, shared by all WGs

  int packThreads = NFRAGS * 32;
  pack_weights_kernel<<<(packThreads + 255) / 256, 256, 0, stream>>>(
      w0, w1, w2, w3, w4, packed);

  int nblocks = (Bsz * Qq) / QPB;       // 8192 workgroups
  liif_mlp_kernel<<<nblocks, 256, 0, stream>>>(
      inp1, inp2, coord, cell1, b0, b1, b2, b3, b4, packed, (float*)d_out);
}
